// Decoder_59012850647323
// MI455X (gfx1250) — matrix operand order repaired and checked by reading, not compiled, not run on hardware
//
#include <hip/hip_runtime.h>

typedef __attribute__((ext_vector_type(16))) _Float16 v16h;
typedef __attribute__((ext_vector_type(8)))  float    v8f;

#define BSZ  16
#define NSEQ 512
#define CDIM 256
#define HDIM 256
#define PDIM 256
#define NP   513
#define G3   768   // 3*H

#define DEV static __device__ __forceinline__

DEV float sigm(float x) { return 1.f / (1.f + __expf(-x)); }

// ---------- WMMA fragment helpers (wave32, 16x16x32 f16 -> f32) ----------
// A: 16x32 (MxK), contiguous k order (matches B fragments):
//  lanes 0-15: M=lane, K=0..15
//  lanes16-31: M=lane-16, K=16..31
DEV v16h frag_A_f16(const _Float16* A, int lda, int k0, int lane) {
  v16h f;
  int m  = lane & 15;
  int kb = k0 + ((lane & 16) ? 16 : 0);
  const _Float16* row = A + (size_t)m * lda;
#pragma unroll
  for (int i = 0; i < 8; ++i) {
    int k = kb + 2 * i;
    f[2 * i]     = row[k];
    f[2 * i + 1] = row[k + 1];
  }
  return f;
}

// B: 32x16 (KxN) with B[k][n] = W[n][k] (W row-major [16 cols x K]).
// lanes 0-15 hold K=0..15 (v_i: K=2i,2i+1), lanes 16-31 hold K=16..31.
DEV v16h frag_B_f16(const _Float16* W, int ldw, int k0, int lane) {
  v16h f;
  int n  = lane & 15;
  int kb = k0 + ((lane & 16) ? 16 : 0);
  const _Float16* row = W + (size_t)n * ldw + kb;
#pragma unroll
  for (int i = 0; i < 8; ++i) { f[2 * i] = row[2 * i]; f[2 * i + 1] = row[2 * i + 1]; }
  return f;
}

// B fragment from f32 feats_p[b] : B[k][n] = (n==0)?0:feats[(b*C+k)*NSEQ + n-1]
DEV v16h frag_B_feats(const float* featsb, int k0, int n0, int lane) {
  v16h f;
  int n  = n0 + (lane & 15);
  int kb = k0 + ((lane & 16) ? 16 : 0);
  bool valid = (n >= 1) && (n <= NSEQ);
  const float* col = featsb + (valid ? (n - 1) : 0);
#pragma unroll
  for (int i = 0; i < 8; ++i) {
    int k = kb + 2 * i;
    f[2 * i]     = valid ? (_Float16)col[(size_t)k * NSEQ]       : (_Float16)0.f;
    f[2 * i + 1] = valid ? (_Float16)col[(size_t)(k + 1) * NSEQ] : (_Float16)0.f;
  }
  return f;
}

DEV v8f wmma32(v16h a, v16h b, v8f c) {
  return __builtin_amdgcn_wmma_f32_16x16x32_f16(false, a, false, b, (short)0, c, false, false);
}

// D (f32 16x16): lane -> N = lane&15 ; VGPR i -> M = i + (lane>=16 ? 8 : 0)
DEV void store_D_bias(float* Y, int ldy, const float* bias, int n0, v8f acc, int lane) {
  int n  = lane & 15;
  int mb = (lane & 16) ? 8 : 0;
  float bv = bias ? bias[n0 + n] : 0.f;
#pragma unroll
  for (int i = 0; i < 8; ++i) Y[(size_t)(mb + i) * ldy + n] = acc[i] + bv;
}

// ---------- conversion / setup kernels ----------
__global__ void cvt_f32_f16(const float* __restrict__ src, _Float16* __restrict__ dst, int n) {
  int i = blockIdx.x * blockDim.x + threadIdx.x;
  if (i < n) dst[i] = (_Float16)src[i];
}

__global__ void cvt_submat(const float* __restrict__ src, _Float16* __restrict__ dst,
                           int rows, int cols, int src_ld, int src_off) {
  int i = blockIdx.x * blockDim.x + threadIdx.x;
  if (i < rows * cols) {
    int r = i / cols, c = i % cols;
    dst[i] = (_Float16)src[(size_t)r * src_ld + src_off + c];
  }
}

// xh[(t*16+b)*C + c] = feats[(b*C+c)*NSEQ + t]   (context at step t)
__global__ void build_xh(const float* __restrict__ feats, _Float16* __restrict__ xh) {
  int i = blockIdx.x * blockDim.x + threadIdx.x;  // NSEQ*BSZ*CDIM
  if (i < NSEQ * BSZ * CDIM) {
    int t = i >> 12;            // / (16*256)
    int b = (i >> 8) & 15;
    int c = i & 255;
    xh[i] = (_Float16)feats[((size_t)b * CDIM + c) * NSEQ + t];
  }
}

// ---------- generic WMMA GEMM: Y[M,N] = X[M,K] * W[N,K]^T + bias[N] ----------
// Each wave computes TWO adjacent 16x16 N-tiles, reusing the A fragment.
__global__ __launch_bounds__(256) void gemm_f16_wT(
    const _Float16* __restrict__ X, const _Float16* __restrict__ W,
    const float* __restrict__ bias, float* __restrict__ Y,
    int Mt, int Ntp /* N-tile PAIRS */, int Kd, int ldx, int ldw) {
  int wave = blockIdx.x * (blockDim.x >> 5) + (threadIdx.x >> 5);
  int lane = threadIdx.x & 31;
  if (wave >= Mt * Ntp) return;
  int mt = wave / Ntp, np = wave % Ntp;
  const _Float16* Xr  = X + (size_t)mt * 16 * ldx;
  const _Float16* Wr0 = W + (size_t)(2 * np) * 16 * ldw;
  const _Float16* Wr1 = Wr0 + (size_t)16 * ldw;
  int m16 = lane & 15;
  v8f acc0 = {}, acc1 = {};
  for (int k0 = 0; k0 < Kd; k0 += 32) {
    if (k0 + 32 < Kd) {  // CDNA5 global_prefetch_b8 of next k-step streams
      __builtin_prefetch(Xr  + (size_t)m16 * ldx + k0 + 32, 0, 1);
      __builtin_prefetch(Wr0 + (size_t)m16 * ldw + k0 + 32, 0, 1);
      __builtin_prefetch(Wr1 + (size_t)m16 * ldw + k0 + 32, 0, 1);
    }
    v16h a = frag_A_f16(Xr, ldx, k0, lane);
    acc0 = wmma32(a, frag_B_f16(Wr0, ldw, k0, lane), acc0);
    acc1 = wmma32(a, frag_B_f16(Wr1, ldw, k0, lane), acc1);
  }
  int Ncols = Ntp * 32;
  float* Yr = Y + (size_t)mt * 16 * Ncols;
  store_D_bias(Yr + (2 * np) * 16,     Ncols, bias, (2 * np) * 16,     acc0, lane);
  store_D_bias(Yr + (2 * np + 1) * 16, Ncols, bias, (2 * np + 1) * 16, acc1, lane);
}

// pf[b,p,n] = mask_p[b,n] * (b_fp[p] + sum_c w_fp[p,c]*feats_p[b,c,n])
__global__ __launch_bounds__(256) void pf_kernel(
    const _Float16* __restrict__ w_fph, const float* __restrict__ feats,
    const float* __restrict__ mask, const float* __restrict__ b_fp,
    float* __restrict__ pf) {
  int wave = blockIdx.x * 8 + (threadIdx.x >> 5);
  int lane = threadIdx.x & 31;
  const int NT = 33;                       // ceil(513/16)
  if (wave >= BSZ * 16 * NT) return;
  int b   = wave / (16 * NT);
  int rem = wave % (16 * NT);
  int pt  = rem / NT, nt = rem % NT;
  const float* fb = feats + (size_t)b * CDIM * NSEQ;
  const _Float16* Aw = w_fph + (size_t)pt * 16 * CDIM;
  int m16 = lane & 15;
  v8f acc = {};
  for (int k0 = 0; k0 < CDIM; k0 += 32) {
    if (k0 + 32 < CDIM)
      __builtin_prefetch(Aw + (size_t)m16 * CDIM + k0 + 32, 0, 1);
    acc = wmma32(frag_A_f16(Aw, CDIM, k0, lane),
                 frag_B_feats(fb, k0, nt * 16, lane), acc);
  }
  int n = nt * 16 + (lane & 15);
  if (n < NP) {
    float mk = (n == 0) ? 1.f : mask[b * NSEQ + n - 1];
    int mb = (lane & 16) ? 8 : 0;
#pragma unroll
    for (int i = 0; i < 8; ++i) {
      int p = pt * 16 + mb + i;
      pf[((size_t)b * PDIM + p) * NP + n] = mk * (acc[i] + b_fp[p]);
    }
  }
}

// fsum[b,c] = sum_j feats[b,c,j]*mask[b,j]; summask[b] = 1 + sum mask; zero saw/csaw
__global__ void fsum_kernel(const float* __restrict__ feats, const float* __restrict__ mask,
                            float* fsum, float* summask, float* saw, float* csaw) {
  int b = blockIdx.x, c = threadIdx.x;
  const float* row = feats + ((size_t)b * CDIM + c) * NSEQ;
  const float* mr  = mask + (size_t)b * NSEQ;
  float s = 0.f;
  for (int j = 0; j < NSEQ; ++j) s += row[j] * mr[j];
  fsum[b * CDIM + c] = s;
  if (c == 0) {
    float sm = 1.f;
    for (int j = 0; j < NSEQ; ++j) sm += mr[j];
    summask[b] = sm;
  }
  saw[b * NP + c] = 0.f;  saw[b * NP + 256 + c] = 0.f;
  csaw[b * NP + c] = 0.f; csaw[b * NP + 256 + c] = 0.f;
  if (c == 0) { saw[b * NP + 512] = 0.f; csaw[b * NP + 512] = 0.f; }
}

__global__ void state0_kernel(const float* __restrict__ w_si, const float* __restrict__ b_si,
                              const float* __restrict__ fsum, const float* __restrict__ summask,
                              float* state) {
  int b = blockIdx.x, h = threadIdx.x;
  float s = 0.f;
  for (int c = 0; c < CDIM; ++c) s += w_si[h * CDIM + c] * fsum[b * CDIM + c];
  state[b * HDIM + h] = (s + b_si[h] * summask[b]) / summask[b];
}

// ---------- sequential step kernels ----------
// S1: gh1 = state@w_hh1^T + b_hh1 (WMMA); GRU1 -> h1; query = h1@w_qh^T + qc_all[t]
__global__ __launch_bounds__(256) void step_gru1(
    const float* __restrict__ state, const _Float16* __restrict__ w_hh1h,
    const float* __restrict__ b_hh1, const float* __restrict__ gi1_all,
    const _Float16* __restrict__ w_qhh, const float* __restrict__ qc_all,
    float* __restrict__ h1_out, float* __restrict__ query_out, int t) {
  extern __shared__ char smem[];
  _Float16* sh_a  = (_Float16*)smem;            // 16x256 f16
  float*    sh_gh = (float*)(smem + 8192);      // 16x768 f32
  int tid = threadIdx.x, lane = tid & 31, wv = tid >> 5;

  for (int i = tid; i < BSZ * HDIM; i += 256) sh_a[i] = (_Float16)state[i];
  __syncthreads();

  for (int nt = wv * 6; nt < wv * 6 + 6; ++nt) {      // 48 N-tiles of 768
    const _Float16* Wr = w_hh1h + (size_t)nt * 16 * HDIM;
    v8f acc = {};
    for (int k0 = 0; k0 < HDIM; k0 += 32)
      acc = wmma32(frag_A_f16(sh_a, HDIM, k0, lane),
                   frag_B_f16(Wr, HDIM, k0, lane), acc);
    store_D_bias(sh_gh + nt * 16, G3, b_hh1, nt * 16, acc, lane);
  }
  __syncthreads();

  const float* gi = gi1_all + (size_t)t * BSZ * G3;
  for (int i = tid; i < BSZ * HDIM; i += 256) {
    int b = i >> 8, j = i & 255;
    float r  = sigm(gi[b * G3 + j]       + sh_gh[b * G3 + j]);
    float z  = sigm(gi[b * G3 + 256 + j] + sh_gh[b * G3 + 256 + j]);
    float nn = tanhf(gi[b * G3 + 512 + j] + r * sh_gh[b * G3 + 512 + j]);
    float h1 = (1.f - z) * nn + z * state[i];
    h1_out[i] = h1;
    sh_a[i]   = (_Float16)h1;
  }
  __syncthreads();

  for (int nt = wv * 2; nt < wv * 2 + 2; ++nt) {      // 16 N-tiles of 256
    v8f acc = {};
    for (int k0 = 0; k0 < HDIM; k0 += 32)
      acc = wmma32(frag_A_f16(sh_a, HDIM, k0, lane),
                   frag_B_f16(w_qhh + (size_t)nt * 16 * HDIM, HDIM, k0, lane), acc);
    int n = nt * 16 + (lane & 15);
    int mb = (lane & 16) ? 8 : 0;
#pragma unroll
    for (int i = 0; i < 8; ++i) {
      int b = mb + i;
      query_out[b * PDIM + n] = acc[i] + qc_all[((size_t)t * BSZ + b) * PDIM + n];
    }
  }
}

// S2: logit[b,n] = b_logit + sum_p w_logit[p]*tanh(pf + query + conv(saw) + conv(csaw))
__global__ __launch_bounds__(256) void step_logit(
    const float* __restrict__ pf, const float* __restrict__ query,
    const float* __restrict__ saw, const float* __restrict__ csaw,
    const float* __restrict__ w_cov, const float* __restrict__ b_cov,
    const float* __restrict__ w_cum, const float* __restrict__ b_cum,
    const float* __restrict__ w_logit, const float* __restrict__ b_logit,
    float* __restrict__ logit, float* __restrict__ out_att, int t) {
  int b = blockIdx.x / 33, chunk = blockIdx.x % 33;
  int tid = threadIdx.x;
  int nl = tid >> 4, pl = tid & 15;
  int n = chunk * 16 + nl;
  float partial = 0.f;
  if (n < NP) {
    float sw[7], cw[7];
#pragma unroll
    for (int k = 0; k < 7; ++k) {
      int q = n + k - 3;
      bool v = (q >= 0) && (q < NP);
      sw[k] = v ? saw[b * NP + q] : 0.f;
      cw[k] = v ? csaw[b * NP + q] : 0.f;
    }
    for (int i = 0; i < 16; ++i) {
      int p = pl + 16 * i;
      float conv = b_cov[p] + b_cum[p];
#pragma unroll
      for (int k = 0; k < 7; ++k) conv += w_cov[p * 7 + k] * sw[k] + w_cum[p * 7 + k] * cw[k];
      float f = pf[((size_t)b * PDIM + p) * NP + n] + query[b * PDIM + p] + conv;
      partial += w_logit[p] * tanhf(f);
    }
  }
  for (int m = 8; m >= 1; m >>= 1) partial += __shfl_xor(partial, m, 16);
  if (pl == 0 && n < NP) {
    float lg = partial + b_logit[0];
    logit[b * NP + n] = lg;
    if (n < NSEQ) out_att[((size_t)t * BSZ + b) * NSEQ + n] = lg;
  }
}

// S3: softmax->att->saw/csaw; argmax->ctx_parent; GRU2 (WMMA); classifier (WMMA); pred
__global__ __launch_bounds__(256) void step_finish(
    const float* __restrict__ logit, const float* __restrict__ mask,
    const float* __restrict__ feats, const _Float16* __restrict__ xh,
    float* __restrict__ saw, float* __restrict__ csaw,
    const float* __restrict__ h1, float* __restrict__ state,
    const _Float16* __restrict__ w_ih2h, const _Float16* __restrict__ w_hh2h,
    const float* __restrict__ b_ih2, const float* __restrict__ b_hh2,
    const _Float16* __restrict__ w_c1h, const float* __restrict__ b_c1,
    const float* __restrict__ w_c2, const float* __restrict__ b_c2,
    float* __restrict__ out_preds, int t) {
  extern __shared__ char smem[];
  float*    sh_g   = (float*)smem;                         // gi2|gh2 : 2*16*768 f32
  _Float16* sh_cls = (_Float16*)(smem + 98304);            // 16x768 f16 (ctx|parent|h2)
  _Float16* sh_h1  = (_Float16*)(smem + 98304 + 24576);    // 16x256 f16
  float*    sh_y1  = (float*)(smem + 98304 + 24576 + 8192);// 16x256 f32
  __shared__ int   sh_idx[16];
  __shared__ float sh_l4[64];
  int tid = threadIdx.x, lane = tid & 31, wv = tid >> 5;

  // 1) per-b softmax + raw argmax (16 threads per b)
  {
    int b = tid >> 4, g = tid & 15;
    const float* lg = logit + b * NP;
    float smax = -3.4e38f, amax = -3.4e38f; int aidx = 0;
    for (int n = g; n < NP; n += 16) {
      float mt = (n <= t) ? ((n == 0) ? 1.f : mask[b * NSEQ + n - 1]) : 0.f;
      float lv = lg[n] - (1.f - mt) * 1e8f;
      if (lv > smax) smax = lv;
      if (n <= t) { float rv = lg[n]; if (rv > amax) { amax = rv; aidx = n; } }
    }
    for (int m = 8; m >= 1; m >>= 1) {
      float o = __shfl_xor(smax, m, 16); if (o > smax) smax = o;
      float oa = __shfl_xor(amax, m, 16); int oi = __shfl_xor(aidx, m, 16);
      if (oa > amax || (oa == amax && oi < aidx)) { amax = oa; aidx = oi; }
    }
    float ssum = 0.f;
    for (int n = g; n < NP; n += 16) {
      float mt = (n <= t) ? ((n == 0) ? 1.f : mask[b * NSEQ + n - 1]) : 0.f;
      ssum += __expf(lg[n] - (1.f - mt) * 1e8f - smax);
    }
    for (int m = 8; m >= 1; m >>= 1) ssum += __shfl_xor(ssum, m, 16);
    float inv = 1.f / ssum;
    for (int n = g; n < NP; n += 16) {
      float mt = (n <= t) ? ((n == 0) ? 1.f : mask[b * NSEQ + n - 1]) : 0.f;
      float a = __expf(lg[n] - (1.f - mt) * 1e8f - smax) * inv;
      saw[b * NP + n] = a;
      csaw[b * NP + n] += a;
    }
    if (g == 0) sh_idx[b] = aidx;
  }
  __syncthreads();

  // 2) cls features: context | ctx_parent | (h2 later); h1 -> f16
  for (int i = tid; i < BSZ * CDIM; i += 256) {
    int b = i >> 8, c = i & 255;
    sh_cls[b * G3 + c] = xh[((size_t)t * BSZ + b) * CDIM + c];
    int idx = sh_idx[b];
    float cp = (idx == 0) ? 0.f : feats[((size_t)b * CDIM + c) * NSEQ + idx - 1];
    sh_cls[b * G3 + 256 + c] = (_Float16)cp;
    sh_h1[i] = (_Float16)h1[i];
  }
  __syncthreads();

  // 3) gi2 = parent@w_ih2^T, gh2 = h1@w_hh2^T (96 tiles, 12/wave)
  for (int tix = wv * 12; tix < wv * 12 + 12; ++tix) {
    int which = tix / 48, nt = tix % 48;
    const _Float16* A   = which ? sh_h1 : (sh_cls + 256);
    int lda             = which ? HDIM : G3;
    const _Float16* W   = (which ? w_hh2h : w_ih2h) + (size_t)nt * 16 * HDIM;
    const float*    bia = which ? b_hh2 : b_ih2;
    v8f acc = {};
    for (int k0 = 0; k0 < HDIM; k0 += 32)
      acc = wmma32(frag_A_f16(A, lda, k0, lane), frag_B_f16(W, HDIM, k0, lane), acc);
    store_D_bias(sh_g + which * (BSZ * G3) + nt * 16, G3, bia, nt * 16, acc, lane);
  }
  __syncthreads();

  // 4) GRU2 -> h2 = new state, also into sh_cls[...,512:768]
  for (int i = tid; i < BSZ * HDIM; i += 256) {
    int b = i >> 8, j = i & 255;
    const float* gi = sh_g + b * G3;
    const float* gh = sh_g + BSZ * G3 + b * G3;
    float r  = sigm(gi[j] + gh[j]);
    float z  = sigm(gi[256 + j] + gh[256 + j]);
    float nn = tanhf(gi[512 + j] + r * gh[512 + j]);
    float h2 = (1.f - z) * nn + z * h1[i];
    state[i] = h2;
    sh_cls[b * G3 + 512 + j] = (_Float16)h2;
  }
  __syncthreads();

  // 5) y1 = tanh(cls_feat@w_c1^T + b_c1)  (16 tiles, K=768)
  for (int nt = wv * 2; nt < wv * 2 + 2; ++nt) {
    v8f acc = {};
    for (int k0 = 0; k0 < G3; k0 += 32)
      acc = wmma32(frag_A_f16(sh_cls, G3, k0, lane),
                   frag_B_f16(w_c1h + (size_t)nt * 16 * G3, G3, k0, lane), acc);
    int n = nt * 16 + (lane & 15);
    int mb = (lane & 16) ? 8 : 0;
#pragma unroll
    for (int i = 0; i < 8; ++i) sh_y1[(mb + i) * HDIM + n] = tanhf(acc[i] + b_c1[n]);
  }
  __syncthreads();

  // 6) 4-way logits + argmax
  if (tid < 64) {
    int b = tid >> 2, v = tid & 3;
    float s = b_c2[v];
    for (int c = 0; c < HDIM; ++c) s += sh_y1[b * HDIM + c] * w_c2[v * HDIM + c];
    sh_l4[tid] = s;
  }
  __syncthreads();
  if (tid < 16) {
    int best = 0; float bv = sh_l4[tid * 4];
    for (int v = 1; v < 4; ++v) { float x = sh_l4[tid * 4 + v]; if (x > bv) { bv = x; best = v; } }
    out_preds[(size_t)t * BSZ + tid] = (float)best;
  }
}

// ---------------------------- host launcher ----------------------------
extern "C" void kernel_launch(void* const* d_in, const int* in_sizes, int n_in,
                              void* d_out, int out_size, void* d_ws, size_t ws_size,
                              hipStream_t stream) {
  (void)in_sizes; (void)n_in; (void)out_size; (void)ws_size;
  const float* feats   = (const float*)d_in[0];
  const float* mask    = (const float*)d_in[1];
  const float* w_fp    = (const float*)d_in[2];
  const float* b_fp    = (const float*)d_in[3];
  const float* w_si    = (const float*)d_in[4];
  const float* b_si    = (const float*)d_in[5];
  const float* w_ih1   = (const float*)d_in[6];
  const float* w_hh1   = (const float*)d_in[7];
  const float* b_ih1   = (const float*)d_in[8];
  const float* b_hh1   = (const float*)d_in[9];
  const float* w_ih2   = (const float*)d_in[10];
  const float* w_hh2   = (const float*)d_in[11];
  const float* b_ih2   = (const float*)d_in[12];
  const float* b_hh2   = (const float*)d_in[13];
  const float* w_q     = (const float*)d_in[14];
  const float* b_q     = (const float*)d_in[15];
  const float* w_cov   = (const float*)d_in[16];
  const float* b_cov   = (const float*)d_in[17];
  const float* w_cum   = (const float*)d_in[18];
  const float* b_cum   = (const float*)d_in[19];
  const float* w_logit = (const float*)d_in[20];
  const float* b_logit = (const float*)d_in[21];
  const float* w_c1    = (const float*)d_in[22];
  const float* b_c1    = (const float*)d_in[23];
  const float* w_c2    = (const float*)d_in[24];
  const float* b_c2    = (const float*)d_in[25];
  float* out = (float*)d_out;

  char* ws = (char*)d_ws;
  size_t off = 0;
  auto alloc = [&](size_t bytes) -> char* {
    char* p = ws + off; off += (bytes + 255) & ~(size_t)255; return p;
  };
  // f16 scratch
  _Float16* xh     = (_Float16*)alloc((size_t)NSEQ * BSZ * CDIM * 2);
  _Float16* w_ih1h = (_Float16*)alloc((size_t)G3 * CDIM * 2);
  _Float16* w_hh1h = (_Float16*)alloc((size_t)G3 * HDIM * 2);
  _Float16* w_ih2h = (_Float16*)alloc((size_t)G3 * CDIM * 2);
  _Float16* w_hh2h = (_Float16*)alloc((size_t)G3 * HDIM * 2);
  _Float16* w_qch  = (_Float16*)alloc((size_t)PDIM * CDIM * 2);
  _Float16* w_qhh  = (_Float16*)alloc((size_t)PDIM * HDIM * 2);
  _Float16* w_c1h  = (_Float16*)alloc((size_t)HDIM * G3 * 2);
  _Float16* w_fph  = (_Float16*)alloc((size_t)PDIM * CDIM * 2);
  // f32 scratch
  float* gi1_all = (float*)alloc((size_t)NSEQ * BSZ * G3 * 4);
  float* qc_all  = (float*)alloc((size_t)NSEQ * BSZ * PDIM * 4);
  float* pf      = (float*)alloc((size_t)BSZ * PDIM * NP * 4);
  float* fsum    = (float*)alloc((size_t)BSZ * CDIM * 4);
  float* summask = (float*)alloc(BSZ * 4);
  float* state   = (float*)alloc((size_t)BSZ * HDIM * 4);
  float* h1      = (float*)alloc((size_t)BSZ * HDIM * 4);
  float* query   = (float*)alloc((size_t)BSZ * PDIM * 4);
  float* saw     = (float*)alloc((size_t)BSZ * NP * 4);
  float* csaw    = (float*)alloc((size_t)BSZ * NP * 4);
  float* logit   = (float*)alloc((size_t)BSZ * NP * 4);

  // --- weight conversion ---
  auto cvt = [&](const float* s, _Float16* d, int n) {
    cvt_f32_f16<<<(n + 255) / 256, 256, 0, stream>>>(s, d, n);
  };
  cvt(w_ih1, w_ih1h, G3 * CDIM);
  cvt(w_hh1, w_hh1h, G3 * HDIM);
  cvt(w_ih2, w_ih2h, G3 * CDIM);
  cvt(w_hh2, w_hh2h, G3 * HDIM);
  cvt(w_c1,  w_c1h,  HDIM * G3);
  cvt(w_fp,  w_fph,  PDIM * CDIM);
  cvt_submat<<<(PDIM * CDIM + 255) / 256, 256, 0, stream>>>(w_q, w_qch, PDIM, CDIM, CDIM + HDIM, 0);
  cvt_submat<<<(PDIM * HDIM + 255) / 256, 256, 0, stream>>>(w_q, w_qhh, PDIM, HDIM, CDIM + HDIM, CDIM);
  build_xh<<<(NSEQ * BSZ * CDIM + 255) / 256, 256, 0, stream>>>(feats, xh);

  // --- big parallel WMMA GEMMs (all-step precompute) ---
  // gi1_all[8192,768] = xh @ w_ih1^T + b_ih1 ; 512 M-tiles x 24 N-tile-pairs
  gemm_f16_wT<<<(512 * 24) / 8, 256, 0, stream>>>(xh, w_ih1h, b_ih1, gi1_all, 512, 24, CDIM, CDIM, CDIM);
  // qc_all[8192,256] = xh @ w_q[:, :C]^T + b_q ; 512 x 8 pairs
  gemm_f16_wT<<<(512 * 8) / 8, 256, 0, stream>>>(xh, w_qch, b_q, qc_all, 512, 8, CDIM, CDIM, CDIM);
  // project_feats
  pf_kernel<<<(BSZ * 16 * 33) / 8, 256, 0, stream>>>(w_fph, feats, mask, b_fp, pf);
  // state0 + zero saw/csaw
  fsum_kernel<<<BSZ, 256, 0, stream>>>(feats, mask, fsum, summask, saw, csaw);
  state0_kernel<<<BSZ, 256, 0, stream>>>(w_si, b_si, fsum, summask, state);

  // --- sequential decode ---
  const size_t S1_SHM = 8192 + (size_t)BSZ * G3 * 4;                     // 57344
  const size_t S3_SHM = 98304 + 24576 + 8192 + (size_t)BSZ * HDIM * 4;   // 147456
  for (int t = 0; t < NSEQ; ++t) {
    step_gru1<<<1, 256, S1_SHM, stream>>>(state, w_hh1h, b_hh1, gi1_all,
                                          w_qhh, qc_all, h1, query, t);
    step_logit<<<BSZ * 33, 256, 0, stream>>>(pf, query, saw, csaw,
                                             w_cov, b_cov, w_cum, b_cum,
                                             w_logit, b_logit, logit,
                                             out + NSEQ * BSZ, t);
    step_finish<<<1, 256, S3_SHM, stream>>>(logit, mask, feats, xh, saw, csaw,
                                            h1, state, w_ih2h, w_hh2h, b_ih2, b_hh2,
                                            w_c1h, b_c1, w_c2, b_c2, out, t);
  }
}
